// MemoryEfficientAttention_65687229825743
// MI455X (gfx1250) — compile-verified, matrix operand order checked by reading, not run on hardware
//
#include <hip/hip_runtime.h>
#include <hip/hip_bf16.h>

// ---------------------------------------------------------------------------
// Types for CDNA5 WMMA (wave32): D(16x16 f32) = A(16x32 f16) * B(32x16 f16) + C
// ---------------------------------------------------------------------------
typedef _Float16 h16;
typedef __attribute__((ext_vector_type(16))) _Float16 v16h;
typedef __attribute__((ext_vector_type(8)))  _Float16 v8h;
typedef __attribute__((ext_vector_type(8)))  float    v8f;

#define Bc   2
#define Tc   4096
#define DMc  2048
#define Hc   16
#define HKVc 4
#define HDc  128
#define BTc  (Bc * Tc)   // 8192

__device__ __forceinline__ v8f wmma_f16(v16h a, v16h b, v8f c) {
  // 8 args: (neg_a, A, neg_b, B, c_mod, C, reuse_a, reuse_b)
  return __builtin_amdgcn_wmma_f32_16x16x32_f16(false, a, false, b, (short)0, c,
                                                false, false);
}

// Fragment loader: per-lane the fragment needs 8 contiguous f16 at +0 (VGPRs
// 0-3) and 8 contiguous f16 at +16 elements (VGPRs 4-7). Two 16B loads.
__device__ __forceinline__ v16h load_frag(const h16* p) {
  v8h lo = *(const v8h*)(p);
  v8h hi = *(const v8h*)(p + 16);
  return __builtin_shufflevector(lo, hi, 0, 1, 2, 3, 4, 5, 6, 7,
                                 8, 9, 10, 11, 12, 13, 14, 15);
}

// ---------------------------------------------------------------------------
// Elementwise f32 -> f16 convert
// ---------------------------------------------------------------------------
__global__ void cvt_f32_f16(const float* __restrict__ in, h16* __restrict__ out,
                            int n) {
  int stride = gridDim.x * blockDim.x;
  for (int i = blockIdx.x * blockDim.x + threadIdx.x; i < n; i += stride)
    out[i] = (h16)in[i];
}

// W (K x N) f32 row-major  ->  Wt (N x K) f16 row-major
__global__ void transpose_cvt(const float* __restrict__ in, h16* __restrict__ out,
                              int K, int N) {
  int n = K * N;
  int stride = gridDim.x * blockDim.x;
  for (int i = blockIdx.x * blockDim.x + threadIdx.x; i < n; i += stride) {
    int k = i / N, c = i - k * N;
    out[(size_t)c * K + k] = (h16)in[i];
  }
}

// ---------------------------------------------------------------------------
// WMMA GEMM: C(M x N) = A(M x K) * W(K x N), with W given transposed (N x K).
// One wave computes a 64x64 tile (4 M-subtiles x 4 N-subtiles): per K-step of
// 32 -> 8 fragment loads (16 b128) feed 16 v_wmma ops (0.5 loads/WMMA).
// WGP-scope speculative prefetch ~8 K-steps ahead warms the near cache.
// OMODE 0: f16 row-major out. OMODE 1: f16 out in (b,kh,d,t) layout (for V).
// OMODE 2: f32 row-major out.
// ---------------------------------------------------------------------------
template <int OMODE>
__global__ __launch_bounds__(256) void gemm_wmma(const h16* __restrict__ A,
                                                 const h16* __restrict__ Bt,
                                                 void* __restrict__ Out,
                                                 int M, int N, int Kd) {
  const int lane  = threadIdx.x & 31;
  const int wave  = (blockIdx.x * blockDim.x + threadIdx.x) >> 5;
  const int mtiles = M >> 6;
  const int mt = wave % mtiles;
  const int nt = wave / mtiles;
  const int m0 = mt << 6;
  const int n0 = nt << 6;
  if (n0 >= N) return;
  const int n16   = lane & 15;
  const int half8 = (lane >> 4) << 3;

  v8f zero = {0.f, 0.f, 0.f, 0.f, 0.f, 0.f, 0.f, 0.f};
  v8f acc[4][4];
#pragma unroll
  for (int i = 0; i < 4; ++i)
#pragma unroll
    for (int j = 0; j < 4; ++j) acc[i][j] = zero;

  const h16* ap[4];
  const h16* bp[4];
#pragma unroll
  for (int i = 0; i < 4; ++i) {
    ap[i] = A + (size_t)(m0 + i * 16 + n16) * Kd + half8;
    bp[i] = Bt + (size_t)(n0 + i * 16 + n16) * Kd + half8;
  }

  for (int kk = 0; kk < Kd; kk += 32) {
    v16h a0 = load_frag(ap[0] + kk);
    v16h a1 = load_frag(ap[1] + kk);
    v16h a2 = load_frag(ap[2] + kk);
    v16h a3 = load_frag(ap[3] + kk);
#pragma unroll
    for (int j = 0; j < 4; ++j) {
      v16h b = load_frag(bp[j] + kk);
      acc[0][j] = wmma_f16(a0, b, acc[0][j]);
      acc[1][j] = wmma_f16(a1, b, acc[1][j]);
      acc[2][j] = wmma_f16(a2, b, acc[2][j]);
      acc[3][j] = wmma_f16(a3, b, acc[3][j]);
    }
    // WGP-scope speculative prefetch ~8 K-steps ahead (silently dropped OOB)
#pragma unroll
    for (int i = 0; i < 4; ++i) {
      __builtin_prefetch(ap[i] + kk + 256, 0, 3);
      __builtin_prefetch(bp[i] + kk + 256, 0, 3);
    }
  }

#pragma unroll
  for (int mi = 0; mi < 4; ++mi)
#pragma unroll
    for (int j = 0; j < 4; ++j)
#pragma unroll
      for (int r = 0; r < 8; ++r) {
        int row = m0 + mi * 16 + r + half8;  // C layout: VGPR r -> M=r(+8)
        int col = n0 + j * 16 + n16;
        float v = acc[mi][j][r];
        if (OMODE == 0) {
          ((h16*)Out)[(size_t)row * N + col] = (h16)v;
        } else if (OMODE == 2) {
          ((float*)Out)[(size_t)row * N + col] = v;
        } else {
          // V transposed store: row = b*T+t, col = kh*HD+d -> Vt[b][kh][d][t]
          int bb = row >> 12, tt = row & (Tc - 1);
          int kh = col >> 7, dd = col & (HDc - 1);
          ((h16*)Out)[(((size_t)bb * HKVc + kh) * HDc + dd) * Tc + tt] = (h16)v;
        }
      }
}

// ---------------------------------------------------------------------------
// Flash attention (causal, GQA). A block's 4 waves own 4 consecutive 16-row
// q-tiles of the same (b,h), so they share every K/V chunk: the block stages
// each 32-key chunk (K: 32x128, V transposed: 128x32, 16KB f16) into LDS once
// (coalesced b128 global loads -> b128 LDS stores), then all waves read WMMA
// fragments from LDS. Per chunk per wave: 8 wmma scores + online softmax in
// registers (shfl_xor) + P transpose via wave-private LDS + 8 wmma for P*V.
// ---------------------------------------------------------------------------
__global__ __launch_bounds__(128) void attn_wmma(const h16* __restrict__ Q,
                                                 const h16* __restrict__ Kb,
                                                 const h16* __restrict__ Vt,
                                                 h16* __restrict__ Ao) {
  __shared__ h16 kbuf[32 * 128];     // K chunk: key-major rows of 128 d
  __shared__ h16 vbuf[128 * 32];     // V chunk transposed: d-major rows of 32 keys
  __shared__ h16 lds_p[4][16 * 32];  // wave-private P tiles (16q x 32k)
  const int tid   = threadIdx.x;
  const int lane  = tid & 31;
  const int wv    = tid >> 5;
  const int gw    = blockIdx.x * 4 + wv;
  const int qt    = gw & 255;          // T/16 tiles; block = 4 consecutive qt
  const int h     = (gw >> 8) & 15;
  const int b     = gw >> 12;
  const int q0    = qt << 4;
  const int kh    = h >> 2;            // N_REP = 4
  const int n16   = lane & 15;
  const int half8 = (lane >> 4) << 3;
  const float scale = 0.08838834764831845f;  // 1/sqrt(128)

  // Q fragments for the whole KV loop (16 rows x 128 d = 4 A-fragments)
  const h16* qrow = Q + (size_t)(b * Tc + q0 + n16) * (Hc * HDc) + h * HDc + half8;
  v16h qfrag[4];
#pragma unroll
  for (int d = 0; d < 4; ++d) qfrag[d] = load_frag(qrow + d * 32);

  v8f zero = {0.f, 0.f, 0.f, 0.f, 0.f, 0.f, 0.f, 0.f};
  v8f o[8];
  float m[8], l[8];
#pragma unroll
  for (int dt = 0; dt < 8; ++dt) o[dt] = zero;
#pragma unroll
  for (int r = 0; r < 8; ++r) { m[r] = -3.0e38f; l[r] = 0.f; }

  const h16* kbase = Kb + (size_t)(b * Tc) * (HKVc * HDc) + kh * HDc;
  const h16* vbase = Vt + ((size_t)b * HKVc + kh) * HDc * (size_t)Tc;
  h16* pbuf = lds_p[wv];

  const int kmax = q0 + 16;  // this wave's exclusive causal limit
  // block-uniform loop bound: causal limit of the block's last q-tile
  const int kmax_blk = ((blockIdx.x * 4 + 3) & 255) * 16 + 16;

  // staging assignments (block-wide, 128 threads)
  const int krow = tid >> 2, kseg = (tid & 3) << 5;  // K: 32 rows x 4 segs of 32
  const h16* gk = kbase + (size_t)krow * (HKVc * HDc) + kseg;
  const h16* gv = vbase + (size_t)tid * Tc;          // V: 128 d-rows x 32 keys
  h16* sk = kbuf + krow * 128 + kseg;
  h16* sv = vbuf + tid * 32;

  for (int ks = 0; ks < kmax_blk; ks += 32) {
    // ---- cooperative stage of K/V chunk into LDS ----
#pragma unroll
    for (int j = 0; j < 4; ++j)
      *(v8h*)(sk + j * 8) = *(const v8h*)(gk + (size_t)ks * (HKVc * HDc) + j * 8);
#pragma unroll
    for (int j = 0; j < 4; ++j)
      *(v8h*)(sv + j * 8) = *(const v8h*)(gv + ks + j * 8);
    // WGP-scope prefetch of next chunk while this one is consumed
    __builtin_prefetch(gk + (size_t)(ks + 32) * (HKVc * HDc), 0, 3);
    __builtin_prefetch(gv + ks + 32, 0, 3);
    __syncthreads();

    if (ks < kmax) {
      // ---- scores: two 16x16 f32 tiles over 128-deep dot products ----
      v8f s0 = zero, s1 = zero;
#pragma unroll
      for (int d = 0; d < 4; ++d) {
        v16h kf0 = load_frag(kbuf + n16 * 128 + d * 32 + half8);
        v16h kf1 = load_frag(kbuf + (16 + n16) * 128 + d * 32 + half8);
        s0 = wmma_f16(qfrag[d], kf0, s0);
        s1 = wmma_f16(qfrag[d], kf1, s1);
      }
      // ---- online softmax, C-layout: VGPR r -> row r+8*half, lane -> col ----
#pragma unroll
      for (int r = 0; r < 8; ++r) {
        int qp = q0 + r + half8;
        int c0 = ks + n16, c1 = c0 + 16;
        float v0 = (c0 <= qp) ? s0[r] * scale : -3.0e38f;
        float v1 = (c1 <= qp) ? s1[r] * scale : -3.0e38f;
        float mx = fmaxf(v0, v1);
#pragma unroll
        for (int off = 1; off < 16; off <<= 1)
          mx = fmaxf(mx, __shfl_xor(mx, off, 32));
        float mn = fmaxf(m[r], mx);
        float p0 = __expf(v0 - mn);
        float p1 = __expf(v1 - mn);
        float rs = p0 + p1;
#pragma unroll
        for (int off = 1; off < 16; off <<= 1) rs += __shfl_xor(rs, off, 32);
        float corr = __expf(m[r] - mn);
        l[r] = l[r] * corr + rs;
        m[r] = mn;
#pragma unroll
        for (int dt = 0; dt < 8; ++dt) o[dt][r] *= corr;
        // spill P row-major into wave-private LDS (transpose staging)
        pbuf[(r + half8) * 32 + n16]      = (h16)p0;
        pbuf[(r + half8) * 32 + n16 + 16] = (h16)p1;
      }
      asm volatile("s_wait_dscnt 0" ::: "memory");
      // reload P in A-fragment layout: lane row n16, cols half8../16+half8..
      v8h plo = *(const v8h*)(pbuf + n16 * 32 + half8);
      v8h phi = *(const v8h*)(pbuf + n16 * 32 + 16 + half8);
      v16h pfrag = __builtin_shufflevector(plo, phi, 0, 1, 2, 3, 4, 5, 6, 7,
                                           8, 9, 10, 11, 12, 13, 14, 15);
      // ---- O += P(16x32) * V(32x128) from transposed-V LDS tile ----
#pragma unroll
      for (int dt = 0; dt < 8; ++dt) {
        v16h vf = load_frag(vbuf + (dt * 16 + n16) * 32 + half8);
        o[dt] = wmma_f16(pfrag, vf, o[dt]);
      }
    }
    __syncthreads();  // protect LDS chunk before next stage
  }

  // ---- normalize and store f16 attention output (B,T,H*HD) ----
#pragma unroll
  for (int r = 0; r < 8; ++r) {
    float inv = 1.0f / l[r];
    int row = q0 + r + half8;
    h16* orow = Ao + (size_t)(b * Tc + row) * (Hc * HDc) + h * HDc;
#pragma unroll
    for (int dt = 0; dt < 8; ++dt)
      orow[dt * 16 + n16] = (h16)(o[dt][r] * inv);
  }
}

// ---------------------------------------------------------------------------
// Host-side pipeline
// ---------------------------------------------------------------------------
extern "C" void kernel_launch(void* const* d_in, const int* in_sizes, int n_in,
                              void* d_out, int out_size, void* d_ws, size_t ws_size,
                              hipStream_t stream) {
  const float* x  = (const float*)d_in[0];
  const float* wq = (const float*)d_in[1];
  const float* wk = (const float*)d_in[2];
  const float* wv = (const float*)d_in[3];
  const float* wo = (const float*)d_in[4];

  char* ws = (char*)d_ws;
  size_t off = 0;
  auto take = [&](size_t bytes) -> char* {
    char* p = ws + off;
    off += (bytes + 255) & ~(size_t)255;
    return p;
  };
  h16* xb  = (h16*)take((size_t)BTc * DMc * 2);
  h16* wqT = (h16*)take((size_t)2048 * 2048 * 2);
  h16* wkT = (h16*)take((size_t)512 * 2048 * 2);
  h16* wvT = (h16*)take((size_t)512 * 2048 * 2);
  h16* woT = (h16*)take((size_t)2048 * 2048 * 2);
  h16* Qb  = (h16*)take((size_t)BTc * 2048 * 2);
  h16* Kb  = (h16*)take((size_t)BTc * 512 * 2);
  h16* Vt  = (h16*)take((size_t)BTc * 512 * 2);
  h16* Ao  = (h16*)take((size_t)BTc * 2048 * 2);

  // 1) precision conversion + weight transposes
  cvt_f32_f16<<<8192, 256, 0, stream>>>(x, xb, BTc * DMc);
  transpose_cvt<<<8192, 256, 0, stream>>>(wq, wqT, DMc, 2048);
  transpose_cvt<<<2048, 256, 0, stream>>>(wk, wkT, DMc, 512);
  transpose_cvt<<<2048, 256, 0, stream>>>(wv, wvT, DMc, 512);
  transpose_cvt<<<8192, 256, 0, stream>>>(wo, woT, 2048, DMc);

  // 2) projections: waves = (M/64)*(N/64); 8 waves / 256-thread block
  gemm_wmma<0><<<512, 256, 0, stream>>>(xb, wqT, Qb, BTc, 2048, DMc);
  gemm_wmma<0><<<128, 256, 0, stream>>>(xb, wkT, Kb, BTc, 512, DMc);
  gemm_wmma<1><<<128, 256, 0, stream>>>(xb, wvT, Vt, BTc, 512, DMc);

  // 3) causal GQA flash attention: B*H*(T/16)=8192 waves, 4 waves/block
  attn_wmma<<<2048, 128, 0, stream>>>(Qb, Kb, Vt, Ao);

  // 4) output projection -> fp32
  gemm_wmma<2><<<512, 256, 0, stream>>>(Ao, woT, (float*)d_out, BTc, 2048, DMc);
}
